// BilinearInterpolator_1202590843026
// MI455X (gfx1250) — matrix operand order repaired and checked by reading, not compiled, not run on hardware
//
#include <hip/hip_runtime.h>
#include <hip/hip_bf16.h>

typedef __attribute__((ext_vector_type(16))) _Float16 v16h;
typedef __attribute__((ext_vector_type(8)))  float    v8f;

struct __align__(16) F4 { float x, y, z, w; };

union HV {
  v16h h;
  unsigned int u[8];
};

__device__ __forceinline__ float leaky(float x) {
  return x >= 0.0f ? x : 0.1f * x;
}

__device__ __forceinline__ unsigned int pack2h(float a, float b) {
  union { _Float16 h[2]; unsigned int u; } t;
  t.h[0] = (_Float16)a;
  t.h[1] = (_Float16)b;
  return t.u;
}

// -------------------------------------------------------------------------
// Kernel 1: img_part[b][pix][ch] = sum_{c<5} w1[ch][c] * x5[b][c][pix]
// x5 channels: 0..2 = image, 3 = row/(H-1), 4 = col/(W-1).
// Stored channel-contiguous so the main kernel loads 16 ch as 4x b128.
// -------------------------------------------------------------------------
__global__ __launch_bounds__(256) void imgpart_kernel(
    const float* __restrict__ image, const float* __restrict__ w1,
    float* __restrict__ img) {
  int idx = blockIdx.x * 256 + threadIdx.x;   // B*1024*64 = 262144 threads
  int ch = idx & 63;
  int p  = (idx >> 6) & 1023;
  int b  = idx >> 16;
  int r = p >> 5, c = p & 31;
  const float* wr = w1 + ch * 7;
  const float* im = image + b * 3 * 1024 + p;
  float v = wr[0] * im[0] + wr[1] * im[1024] + wr[2] * im[2048]
          + wr[3] * ((float)r * (1.0f / 31.0f))
          + wr[4] * ((float)c * (1.0f / 31.0f));
  img[idx] = v;   // idx == (b*1024 + p)*64 + ch
}

// -------------------------------------------------------------------------
// Kernel 2: one workgroup per (b, n). 8 wave32s; each wave runs 8 tiles of
// 16 pixels through 4 x (64x64 GEMM + leaky) via v_wmma_f32_16x16x32_f16,
// then mean-pool + 3x64 sigmoid head. Layer loop kept rolled and A-operands
// loaded at point of use to stay under 256 VGPRs (no s_set_vgpr_msb churn;
// >=4 waves/SIMD so cross-wave scheduling hides the ds->wmma latency).
// -------------------------------------------------------------------------
__global__ __launch_bounds__(256) void coordmlp_kernel(
    const float* __restrict__ img,    const float* __restrict__ coords,
    const float* __restrict__ w1,     const float* __restrict__ b1,
    const float* __restrict__ w2,     const float* __restrict__ b2,
    const float* __restrict__ w3,     const float* __restrict__ b3,
    const float* __restrict__ w4,     const float* __restrict__ b4,
    const float* __restrict__ w5,     const float* __restrict__ b5,
    const float* __restrict__ wl,     const float* __restrict__ bl,
    float* __restrict__ out) {
  __shared__ __align__(16) float s_actbias[64];               // crd_part + b1
  __shared__ __align__(16) float s_bias[4][64];               // b2..b5
  __shared__ __align__(16) unsigned int   s_w[4][4][2][32][8];// A-layout f16 weights
  __shared__ __align__(16) unsigned short s_h[8][16][72];     // per-wave h[pix][ch], padded
  __shared__ float s_partial[8][64];
  __shared__ float s_pool[64];

  const int tid = threadIdx.x;
  const int n = blockIdx.x & 127;
  const int b = blockIdx.x >> 7;

  // --- per-(b,n) layer-1 additive term: w1[:,5:7].coords + b1 ---
  if (tid < 64) {
    float cx = coords[(b * 128 + n) * 2 + 0];
    float cy = coords[(b * 128 + n) * 2 + 1];
    s_actbias[tid] = w1[tid * 7 + 5] * cx + w1[tid * 7 + 6] * cy + b1[tid];
  }
  // --- biases for the 4 GEMM layers ---
  {
    int li = tid >> 6, ch = tid & 63;
    const float* bp = (li == 0) ? b2 : (li == 1) ? b3 : (li == 2) ? b4 : b5;
    s_bias[li][ch] = bp[ch];
  }
  // --- swizzle w2..w5 (fp32 row-major [out][in]) into WMMA A-matrix f16 layout ---
  {
    const float* Ws[4] = {w2, w3, w4, w5};
    for (int e = tid; e < 8192; e += 256) {
      int v    = e & 7;
      int lane = (e >> 3) & 31;
      int kc   = (e >> 8) & 1;
      int mt   = (e >> 9) & 3;
      int li   = e >> 11;
      int M  = mt * 16 + (lane & 15);
      int kh = lane >> 4;
      int k  = kc * 32 + kh * 16 + v * 2;
      const float* W = Ws[li];
      s_w[li][mt][kc][lane][v] = pack2h(W[M * 64 + k], W[M * 64 + k + 1]);
    }
  }
  __syncthreads();

  const int wave   = tid >> 5;
  const int lane   = tid & 31;
  const int laneN  = lane & 15;   // pixel within tile (B/D column)
  const int hiHalf = lane >> 4;

  const v8f zf = {0.f, 0.f, 0.f, 0.f, 0.f, 0.f, 0.f, 0.f};
  v8f pacc[4] = {zf, zf, zf, zf};   // pooled-sum accumulator

  for (int t = 0; t < 8; ++t) {
    const int p = (wave * 8 + t) * 16 + laneN;   // this lane's pixel

    // ---- layer 1: build h1 directly in WMMA B layout ----
    HV bmat[2];
#pragma unroll
    for (int kc = 0; kc < 2; ++kc) {
      const int ch0 = kc * 32 + hiHalf * 16;
      const F4* gp =
          reinterpret_cast<const F4*>(img + (size_t)(b * 1024 + p) * 64 + ch0);
      const F4* abp = reinterpret_cast<const F4*>(&s_actbias[ch0]);
#pragma unroll
      for (int j = 0; j < 4; ++j) {
        F4 f = gp[j];
        F4 a = abp[j];
        float x0 = leaky(f.x + a.x);
        float x1 = leaky(f.y + a.y);
        float x2 = leaky(f.z + a.z);
        float x3 = leaky(f.w + a.w);
        bmat[kc].u[j * 2 + 0] = pack2h(x0, x1);
        bmat[kc].u[j * 2 + 1] = pack2h(x2, x3);
      }
    }

    // ---- layers 2..4: WMMA GEMMs with LDS bounce (rolled loop) ----
#pragma unroll 1
    for (int li = 0; li < 3; ++li) {
      v8f c[4] = {zf, zf, zf, zf};
#pragma unroll
      for (int kc = 0; kc < 2; ++kc) {
#pragma unroll
        for (int mt = 0; mt < 4; ++mt) {
          HV a;
          const unsigned int* wp = &s_w[li][mt][kc][lane][0];
#pragma unroll
          for (int d = 0; d < 8; ++d) a.u[d] = wp[d];
          c[mt] = __builtin_amdgcn_wmma_f32_16x16x32_f16(
              false, a.h, false, bmat[kc].h, (short)0, c[mt], false, false);
        }
      }
      // bias + leaky; D layout: lane holds channels chb..chb+7 for pixel laneN
#pragma unroll
      for (int mt = 0; mt < 4; ++mt) {
        const int chb = mt * 16 + hiHalf * 8;
        unsigned int* hp =
            reinterpret_cast<unsigned int*>(&s_h[wave][laneN][chb]);
#pragma unroll
        for (int v2 = 0; v2 < 4; ++v2) {
          float d0 = leaky(c[mt][2 * v2 + 0] + s_bias[li][chb + 2 * v2 + 0]);
          float d1 = leaky(c[mt][2 * v2 + 1] + s_bias[li][chb + 2 * v2 + 1]);
          hp[v2] = pack2h(d0, d1);
        }
      }
      // read back as next layer's B operand (16 contiguous halves per chunk)
#pragma unroll
      for (int kc = 0; kc < 2; ++kc) {
        const unsigned int* rp = reinterpret_cast<const unsigned int*>(
            &s_h[wave][laneN][kc * 32 + hiHalf * 16]);
#pragma unroll
        for (int d = 0; d < 8; ++d) bmat[kc].u[d] = rp[d];
      }
    }

    // ---- layer 5 (w5): WMMA + activation feeds the mean-pool ----
    {
      v8f c[4] = {zf, zf, zf, zf};
#pragma unroll
      for (int kc = 0; kc < 2; ++kc) {
#pragma unroll
        for (int mt = 0; mt < 4; ++mt) {
          HV a;
          const unsigned int* wp = &s_w[3][mt][kc][lane][0];
#pragma unroll
          for (int d = 0; d < 8; ++d) a.u[d] = wp[d];
          c[mt] = __builtin_amdgcn_wmma_f32_16x16x32_f16(
              false, a.h, false, bmat[kc].h, (short)0, c[mt], false, false);
        }
      }
#pragma unroll
      for (int mt = 0; mt < 4; ++mt) {
        const int chb = mt * 16 + hiHalf * 8;
#pragma unroll
        for (int v = 0; v < 8; ++v)
          pacc[mt][v] += leaky(c[mt][v] + s_bias[3][chb + v]);
      }
    }
  }

  // ---- reduce pooled sums across the 16 pixel-lanes of each half-wave ----
#pragma unroll
  for (int mt = 0; mt < 4; ++mt)
#pragma unroll
    for (int v = 0; v < 8; ++v) {
      float x = pacc[mt][v];
      x += __shfl_xor(x, 1, 32);
      x += __shfl_xor(x, 2, 32);
      x += __shfl_xor(x, 4, 32);
      x += __shfl_xor(x, 8, 32);
      pacc[mt][v] = x;
    }
  if (laneN == 0) {
#pragma unroll
    for (int mt = 0; mt < 4; ++mt)
#pragma unroll
      for (int v = 0; v < 8; ++v)
        s_partial[wave][mt * 16 + hiHalf * 8 + v] = pacc[mt][v];
  }
  __syncthreads();

  // deterministic cross-wave combine + mean
  if (tid < 64) {
    float s = 0.f;
#pragma unroll
    for (int w = 0; w < 8; ++w) s += s_partial[w][tid];
    s_pool[tid] = s * (1.0f / 1024.0f);
  }
  __syncthreads();

  // final 3x64 head + sigmoid; out[b][o][n]
  if (tid < 3) {
    float acc = bl[tid];
#pragma unroll
    for (int c = 0; c < 64; ++c) acc += wl[tid * 64 + c] * s_pool[c];
    out[(b * 3 + tid) * 128 + n] = 1.0f / (1.0f + expf(-acc));
  }
}

extern "C" void kernel_launch(void* const* d_in, const int* in_sizes, int n_in,
                              void* d_out, int out_size, void* d_ws, size_t ws_size,
                              hipStream_t stream) {
  const float* image  = (const float*)d_in[0];
  const float* coords = (const float*)d_in[1];
  const float* w1 = (const float*)d_in[2];
  const float* b1 = (const float*)d_in[3];
  const float* w2 = (const float*)d_in[4];
  const float* b2 = (const float*)d_in[5];
  const float* w3 = (const float*)d_in[6];
  const float* b3 = (const float*)d_in[7];
  const float* w4 = (const float*)d_in[8];
  const float* b4 = (const float*)d_in[9];
  const float* w5 = (const float*)d_in[10];
  const float* b5 = (const float*)d_in[11];
  const float* wl = (const float*)d_in[12];
  const float* bl = (const float*)d_in[13];
  float* img = (float*)d_ws;               // B*1024*64 fp32 = 1 MB scratch

  imgpart_kernel<<<(4 * 1024 * 64) / 256, 256, 0, stream>>>(image, w1, img);
  coordmlp_kernel<<<4 * 128, 256, 0, stream>>>(
      img, coords, w1, b1, w2, b2, w3, b3, w4, b4, w5, b5, wl, bl,
      (float*)d_out);

  (void)in_sizes; (void)n_in; (void)out_size; (void)ws_size;
}
